// RelativeSelfAttention_53798760350184
// MI455X (gfx1250) — matrix operand order repaired and checked by reading, not compiled, not run on hardware
//
#include <hip/hip_runtime.h>
#include <hip/hip_bf16.h>
#include <math.h>

// ---------------------------------------------------------------------------
// Relative self-attention for MI455X (gfx1250, wave32, WMMA).
// All matmuls run through V_WMMA_F32_16X16X32_BF16 (fp32 accumulate).
// GEMMs use 32x64 per-wave register blocking: each B fragment feeds 2 WMMAs.
// ---------------------------------------------------------------------------

#define DEVINL __device__ __forceinline__

typedef __attribute__((ext_vector_type(8)))  __bf16 v8bf;
typedef __attribute__((ext_vector_type(16))) __bf16 v16bf;
typedef __attribute__((ext_vector_type(8)))  float  v8f;

constexpr int Bc = 4;      // batch
constexpr int Lc = 1024;   // seq len
constexpr int Ec = 1024;   // embed
constexpr int Hc = 16;     // heads
constexpr int Dc = 64;     // head dim
constexpr int RROWS = 129; // 2K+1
constexpr int RPAD  = 144; // padded to 9 tiles of 16

// round-to-nearest-even fp32 -> bf16 (as raw u16)
DEVINL unsigned short f2bf(float f) {
  union { float f; unsigned u; } x; x.f = f;
  unsigned r = x.u + 0x7FFFu + ((x.u >> 16) & 1u);
  return (unsigned short)(r >> 16);
}

// A fragment: 16x32 bf16, row-major source (lane&15 = row M).
// Per ISA: lanes 0-15 hold K=0..7 (v0-3) and K=16..23 (v4-7);
//          lanes 16-31 hold K=8..15 and K=24..31.
DEVINL v16bf load_a_frag(const unsigned short* __restrict__ src, int ld, int lane) {
  const int r = lane & 15, half = lane >> 4;
  const v8bf lo = *reinterpret_cast<const v8bf*>(src + (size_t)r * ld + half * 8);
  const v8bf hi = *reinterpret_cast<const v8bf*>(src + (size_t)r * ld + 16 + half * 8);
  return __builtin_shufflevector(lo, hi, 0,1,2,3,4,5,6,7,8,9,10,11,12,13,14,15);
}

// B fragment: 32x16 bf16 where column n is contiguous in memory ("B^T" rows).
// Same K order as A: lanes 0-15 hold K=0..7 (v0-3) and K=16..23 (v4-7);
// lanes 16-31 hold K=8..15 and K=24..31.
DEVINL v16bf load_b_frag(const unsigned short* __restrict__ srcT, int ld, int lane) {
  const int n = lane & 15, half = lane >> 4;
  const v8bf lo = *reinterpret_cast<const v8bf*>(srcT + (size_t)n * ld + half * 8);
  const v8bf hi = *reinterpret_cast<const v8bf*>(srcT + (size_t)n * ld + 16 + half * 8);
  return __builtin_shufflevector(lo, hi, 0,1,2,3,4,5,6,7,8,9,10,11,12,13,14,15);
}

DEVINL v8f wmma_bf16(v16bf a, v16bf b, v8f c) {
  return __builtin_amdgcn_wmma_f32_16x16x32_bf16(
      /*neg_a=*/false, a, /*neg_b=*/false, b,
      /*c_mod=*/(short)0, c, /*reuse_a=*/false, /*reuse_b=*/false);
}

// ---------------------------------------------------------------------------
// fp32 -> bf16 conversion kernels
// ---------------------------------------------------------------------------
__global__ void cvt_bf16_kernel(const float* __restrict__ src,
                                unsigned short* __restrict__ dst, int n) {
  for (int i = blockIdx.x * blockDim.x + threadIdx.x; i < n; i += gridDim.x * blockDim.x)
    dst[i] = f2bf(src[i]);
}

__global__ void cvt_relpe_kernel(const float* __restrict__ src,
                                 unsigned short* __restrict__ dst) {
  int i = blockIdx.x * blockDim.x + threadIdx.x;
  if (i < RPAD * Dc) {
    int r = i >> 6;
    dst[i] = (r < RROWS) ? f2bf(src[i]) : (unsigned short)0;
  }
}

// ---------------------------------------------------------------------------
// QKV projection: C(4096x3072) = q_bf(4096x1024) @ W^T, +bias, routed to
//   qh  (B,H,L,D) bf16, pre-scaled by 1/sqrt(D)
//   kh  (B,H,L,D) bf16
//   vhT (B,H,D,L) bf16   (transposed so attn@V B-fragments are contiguous)
// grid (16,48): 256 rows x 64 cols per workgroup; wave w owns 32 rows.
// Register blocking: 2 M-tiles x 4 N-tiles per wave -> each B fragment
// feeds two WMMAs (8 wmma per 12 b128 loads per K-chunk).
// ---------------------------------------------------------------------------
__global__ void qkv_gemm_kernel(const unsigned short* __restrict__ Abf,
                                const unsigned short* __restrict__ Wbf,
                                const float* __restrict__ bias,
                                unsigned short* __restrict__ qh,
                                unsigned short* __restrict__ kh,
                                unsigned short* __restrict__ vhT) {
  const int lane = threadIdx.x & 31;
  const int w    = threadIdx.x >> 5;
  const int m0   = blockIdx.x * 256 + w * 32;
  const int n0   = blockIdx.y * 64;

  const v8f vz = {0.f,0.f,0.f,0.f,0.f,0.f,0.f,0.f};
  v8f acc[2][4] = {{vz, vz, vz, vz}, {vz, vz, vz, vz}};

  const unsigned short* Arow0 = Abf + (size_t)m0 * Ec;
  const unsigned short* Arow1 = Abf + (size_t)(m0 + 16) * Ec;
  for (int kc = 0; kc < Ec; kc += 32) {
    if (kc + 160 <= Ec) {
      __builtin_prefetch(Arow0 + kc + 128, 0, 1);
      __builtin_prefetch(Arow1 + kc + 128, 0, 1);
    }
    v16bf a0 = load_a_frag(Arow0 + kc, Ec, lane);
    v16bf a1 = load_a_frag(Arow1 + kc, Ec, lane);
#pragma unroll
    for (int jt = 0; jt < 4; ++jt) {
      v16bf b = load_b_frag(Wbf + (size_t)(n0 + 16 * jt) * Ec + kc, Ec, lane);
      acc[0][jt] = wmma_bf16(a0, b, acc[0][jt]);
      acc[1][jt] = wmma_bf16(a1, b, acc[1][jt]);
    }
  }

  const int half = lane >> 4, nn = lane & 15;
  const int seg = n0 >> 10; // 0=q, 1=k, 2=v (uniform per workgroup)
#pragma unroll
  for (int mt = 0; mt < 2; ++mt) {
#pragma unroll
    for (int jt = 0; jt < 4; ++jt) {
#pragma unroll
      for (int e = 0; e < 8; ++e) {
        int m = m0 + mt * 16 + e + 8 * half;
        int n = n0 + 16 * jt + nn;
        float v = acc[mt][jt][e] + bias[n];
        int b  = m >> 10, l = m & 1023;
        int nc = n & 1023;
        int h  = nc >> 6, d = nc & 63;
        size_t bh = (size_t)b * Hc + h;
        if (seg == 0)      qh [(bh * Lc + l) * Dc + d] = f2bf(v * 0.125f); // 1/sqrt(64)
        else if (seg == 1) kh [(bh * Lc + l) * Dc + d] = f2bf(v);
        else               vhT[(bh * Dc + d) * Lc + l] = f2bf(v);
      }
    }
  }
}

// ---------------------------------------------------------------------------
// Attention: one workgroup per (b,h, 16-row query tile). 8 waves.
// LDS: bias_r fp32 [16][144] | scores fp32 [16][1024] (reused as partials)
//      | P bf16 [16][1024]
// ---------------------------------------------------------------------------
constexpr int SMEM_BIAS   = 16 * RPAD * 4;            // 9216
constexpr int SMEM_SCORES = 16 * Lc * 4;              // 65536
constexpr int SMEM_P      = 16 * Lc * 2;              // 32768
constexpr int SMEM_TOTAL  = SMEM_BIAS + SMEM_SCORES + SMEM_P;

__global__ void attn_kernel(const unsigned short* __restrict__ qh,
                            const unsigned short* __restrict__ kh,
                            const unsigned short* __restrict__ vhT,
                            const unsigned short* __restrict__ relpe,
                            unsigned short* __restrict__ attn_bf) {
  extern __shared__ char smem[];
  float*          bias_lds  = reinterpret_cast<float*>(smem);
  float*          score_lds = reinterpret_cast<float*>(smem + SMEM_BIAS);
  unsigned short* p_lds     = reinterpret_cast<unsigned short*>(smem + SMEM_BIAS + SMEM_SCORES);
  float*          part      = score_lds; // reused after softmax

  const int tid  = threadIdx.x;
  const int lane = tid & 31, w = tid >> 5;
  const int half = lane >> 4, nn = lane & 15;
  const int bh   = blockIdx.x >> 6;   // 0..63 = b*16+h
  const int i0   = (blockIdx.x & 63) * 16;

  // Query tile A-fragments (pre-scaled bf16), K = 0..31 and 32..63
  const unsigned short* qbase = qh + ((size_t)bh * Lc + i0) * Dc;
  const v16bf qa0 = load_a_frag(qbase,      Dc, lane);
  const v16bf qa1 = load_a_frag(qbase + 32, Dc, lane);

  const v8f vz = {0.f,0.f,0.f,0.f,0.f,0.f,0.f,0.f};

  // ---- Phase 1: bias_r[i, r] = q_scaled[i] . rel_pe[r], 9 column tiles ----
  for (int t = w; t < 9; t += 8) {
    int r0 = t * 16;
    v16bf b0 = load_b_frag(relpe + (size_t)r0 * Dc,      Dc, lane);
    v16bf b1 = load_b_frag(relpe + (size_t)r0 * Dc + 32, Dc, lane);
    v8f c = vz;
    c = wmma_bf16(qa0, b0, c);
    c = wmma_bf16(qa1, b1, c);
#pragma unroll
    for (int e = 0; e < 8; ++e)
      bias_lds[(e + 8 * half) * RPAD + r0 + nn] = c[e];
  }
  __syncthreads();

  // ---- Phase 2: scores + gathered relative bias -> LDS -------------------
  for (int t = 0; t < 8; ++t) {
    int j0 = (w + 8 * t) * 16;
    const unsigned short* kbase = kh + ((size_t)bh * Lc + j0) * Dc;
    v16bf kb0 = load_b_frag(kbase,      Dc, lane);
    v16bf kb1 = load_b_frag(kbase + 32, Dc, lane);
    v8f c = vz;
    c = wmma_bf16(qa0, kb0, c);
    c = wmma_bf16(qa1, kb1, c);
#pragma unroll
    for (int e = 0; e < 8; ++e) {
      int r  = e + 8 * half;     // local row
      int gj = j0 + nn;          // key index
      int rel = gj - (i0 + r);
      rel = rel < -64 ? -64 : (rel > 64 ? 64 : rel);
      score_lds[r * Lc + gj] = c[e] + bias_lds[r * RPAD + rel + 64];
    }
  }
  __syncthreads();

  // ---- Phase 3: row softmax (wave w owns rows 2w, 2w+1) ------------------
#pragma unroll
  for (int rr = 0; rr < 2; ++rr) {
    int r = 2 * w + rr;
    float* row = score_lds + r * Lc;
    float m = -3.0e38f;
    for (int c0 = lane; c0 < Lc; c0 += 32) m = fmaxf(m, row[c0]);
#pragma unroll
    for (int s = 16; s > 0; s >>= 1) m = fmaxf(m, __shfl_xor(m, s, 32));
    float sum = 0.f;
    for (int c0 = lane; c0 < Lc; c0 += 32) {
      float ev = __expf(row[c0] - m);
      row[c0] = ev;
      sum += ev;
    }
#pragma unroll
    for (int s = 16; s > 0; s >>= 1) sum += __shfl_xor(sum, s, 32);
    float inv = 1.0f / sum;
    unsigned short* prow = p_lds + r * Lc;
    for (int c0 = lane; c0 < Lc; c0 += 32) prow[c0] = f2bf(row[c0] * inv);
  }
  __syncthreads();

  // ---- Phase 4: O = P @ V, K-range split across waves --------------------
  v8f o[4] = {vz, vz, vz, vz};
  const int jb = w * 128;
  for (int kc = 0; kc < 128; kc += 32) {
    v16bf pa = load_a_frag(p_lds + jb + kc, Lc, lane);
#pragma unroll
    for (int dt = 0; dt < 4; ++dt) {
      const unsigned short* vrow = vhT + ((size_t)bh * Dc + dt * 16) * Lc + jb + kc;
      v16bf vb = load_b_frag(vrow, Lc, lane);
      o[dt] = wmma_bf16(pa, vb, o[dt]);
    }
  }
  // partial sums into (reused) score region, then sum the 8 wave partials
#pragma unroll
  for (int dt = 0; dt < 4; ++dt)
#pragma unroll
    for (int e = 0; e < 8; ++e)
      part[w * 1024 + (e + 8 * half) * Dc + dt * 16 + nn] = o[dt][e];
  __syncthreads();

  const int b = bh >> 4, h = bh & 15;
  for (int idx = tid; idx < 16 * Dc; idx += 256) {
    float s = 0.f;
#pragma unroll
    for (int ww = 0; ww < 8; ++ww) s += part[ww * 1024 + idx];
    int r = idx >> 6, d = idx & 63;
    attn_bf[((size_t)b * Lc + i0 + r) * Ec + h * Dc + d] = f2bf(s);
  }
}

// ---------------------------------------------------------------------------
// Output projection: out(4096x1024) = attn_bf @ out_w^T + out_b (fp32 out)
// Same 32x64 per-wave register blocking as the QKV GEMM.
// ---------------------------------------------------------------------------
__global__ void out_gemm_kernel(const unsigned short* __restrict__ Abf,
                                const unsigned short* __restrict__ Wbf,
                                const float* __restrict__ bias,
                                float* __restrict__ out) {
  const int lane = threadIdx.x & 31;
  const int w    = threadIdx.x >> 5;
  const int m0   = blockIdx.x * 256 + w * 32;
  const int n0   = blockIdx.y * 64;

  const v8f vz = {0.f,0.f,0.f,0.f,0.f,0.f,0.f,0.f};
  v8f acc[2][4] = {{vz, vz, vz, vz}, {vz, vz, vz, vz}};

  const unsigned short* Arow0 = Abf + (size_t)m0 * Ec;
  const unsigned short* Arow1 = Abf + (size_t)(m0 + 16) * Ec;
  for (int kc = 0; kc < Ec; kc += 32) {
    if (kc + 160 <= Ec) {
      __builtin_prefetch(Arow0 + kc + 128, 0, 1);
      __builtin_prefetch(Arow1 + kc + 128, 0, 1);
    }
    v16bf a0 = load_a_frag(Arow0 + kc, Ec, lane);
    v16bf a1 = load_a_frag(Arow1 + kc, Ec, lane);
#pragma unroll
    for (int jt = 0; jt < 4; ++jt) {
      v16bf b = load_b_frag(Wbf + (size_t)(n0 + 16 * jt) * Ec + kc, Ec, lane);
      acc[0][jt] = wmma_bf16(a0, b, acc[0][jt]);
      acc[1][jt] = wmma_bf16(a1, b, acc[1][jt]);
    }
  }

  const int half = lane >> 4, nn = lane & 15;
#pragma unroll
  for (int mt = 0; mt < 2; ++mt)
#pragma unroll
    for (int jt = 0; jt < 4; ++jt)
#pragma unroll
      for (int e = 0; e < 8; ++e) {
        int m = m0 + mt * 16 + e + 8 * half;
        int n = n0 + 16 * jt + nn;
        out[(size_t)m * Ec + n] = acc[mt][jt][e] + bias[n];
      }
}

// ---------------------------------------------------------------------------
// Host launcher
// ---------------------------------------------------------------------------
extern "C" void kernel_launch(void* const* d_in, const int* in_sizes, int n_in,
                              void* d_out, int out_size, void* d_ws, size_t ws_size,
                              hipStream_t stream) {
  const float* q     = (const float*)d_in[0];
  const float* w_in  = (const float*)d_in[1];
  const float* b_in  = (const float*)d_in[2];
  const float* w_out = (const float*)d_in[3];
  const float* b_out = (const float*)d_in[4];
  const float* relpe = (const float*)d_in[5];
  float* out = (float*)d_out;

  char* ws = (char*)d_ws;
  size_t off = 0;
  auto alloc = [&](size_t bytes) -> unsigned short* {
    unsigned short* p = (unsigned short*)(ws + off);
    off += (bytes + 255) & ~(size_t)255;
    return p;
  };
  const size_t NQ = (size_t)Bc * Lc * Ec;           // 4M elems
  unsigned short* qbf    = alloc(NQ * 2);           // q bf16
  unsigned short* winbf  = alloc((size_t)3 * Ec * Ec * 2);
  unsigned short* woutbf = alloc((size_t)Ec * Ec * 2);
  unsigned short* relbf  = alloc((size_t)RPAD * Dc * 2);
  unsigned short* qh     = alloc(NQ * 2);           // (B,H,L,D) scaled
  unsigned short* kh     = alloc(NQ * 2);           // (B,H,L,D)
  unsigned short* vhT    = alloc(NQ * 2);           // (B,H,D,L)
  unsigned short* attnbf = alloc(NQ * 2);           // (B,L,E)

  cvt_bf16_kernel<<<2048, 256, 0, stream>>>(q,     qbf,    (int)NQ);
  cvt_bf16_kernel<<<2048, 256, 0, stream>>>(w_in,  winbf,  3 * Ec * Ec);
  cvt_bf16_kernel<<<1024, 256, 0, stream>>>(w_out, woutbf, Ec * Ec);
  cvt_relpe_kernel<<<(RPAD * Dc + 255) / 256, 256, 0, stream>>>(relpe, relbf);

  qkv_gemm_kernel<<<dim3(16, 48), 256, 0, stream>>>(qbf, winbf, b_in, qh, kh, vhT);

  attn_kernel<<<dim3(Bc * Hc * (Lc / 16)), 256, SMEM_TOTAL, stream>>>(
      qh, kh, vhT, relbf, attnbf);

  out_gemm_kernel<<<dim3(16, 16), 256, 0, stream>>>(attnbf, woutbf, b_out, out);
}
